// S2SModel_78993038508736
// MI455X (gfx1250) — matrix operand order repaired and checked by reading, compiled, not run on hardware
//
#include <hip/hip_runtime.h>
#include <hip/hip_bf16.h>
#include <hip/hip_fp16.h>
#include <stdint.h>

#define N_NODES 50000
#define N_EDGES 400000
#define HID 128
#define NLAYERS 4
#define CUTOFF 5.0f

typedef __attribute__((ext_vector_type(16))) _Float16 v16h;
typedef __attribute__((ext_vector_type(8)))  float    v8f;

// ---------------- WMMA helpers (A and B both use contiguous k = 16*(lane/16) + s) ----

__device__ __forceinline__ v8f wmma16(v16h a, v16h b, v8f c) {
  // D = A(16x32 f16) x B(32x16 f16) + C(16x16 f32)
  return __builtin_amdgcn_wmma_f32_16x16x32_f16(
      false, a, false, b, (short)0, c, false, false);
}

// A fragment: row-major f16 tile [16][ld], K-slice already offset into 'a'.
__device__ __forceinline__ v16h frag_a(const _Float16* a, int ld, int lane) {
  const int row = lane & 15;
  const int hh  = lane >> 4;
  const _Float16* p = a + row * ld;
  v16h r;
#pragma unroll
  for (int v = 0; v < 8; ++v) {
    const int k = (hh << 4) + (v << 1);
    r[2 * v]     = p[k];
    r[2 * v + 1] = p[k + 1];
  }
  return r;
}

// B fragment from weight packed transposed as wt[n*ldk + k] (n = output col).
// 'wt' pre-offset to (ntile*16)*ldk + ktile*32. 16 contiguous halves per lane.
__device__ __forceinline__ v16h frag_b(const _Float16* wt, int ldk, int lane) {
  const int col = lane & 15;
  const int hh  = lane >> 4;
  const _Float16* p = wt + col * ldk + (hh << 4);
  v16h r;
#pragma unroll
  for (int v = 0; v < 8; ++v) {
    r[2 * v]     = p[2 * v];
    r[2 * v + 1] = p[2 * v + 1];
  }
  return r;
}

// Preload all K-step A fragments for one 16-row tile (reused across N-tiles).
template <int KT>
__device__ __forceinline__ void load_afrags(v16h* af, const _Float16* a, int ld,
                                            int lane) {
#pragma unroll
  for (int kt = 0; kt < KT; ++kt) af[kt] = frag_a(a + kt * 32, ld, lane);
}

// One 16x16 output tile: sum over KT k-steps against packed-transposed weights.
template <int KT>
__device__ __forceinline__ v8f gemm_tile(const v16h* af, const _Float16* wt,
                                         int ldk, int lane) {
  v8f acc = {};
#pragma unroll
  for (int kt = 0; kt < KT; ++kt)
    acc = wmma16(af[kt], frag_b(wt + kt * 32, ldk, lane), acc);
  return acc;
}

__device__ __forceinline__ float silu_f(float x) {
  // x * sigmoid(x) with hardware-rate v_rcp_f32 (1-ulp) instead of IEEE div
  return x * __builtin_amdgcn_rcpf(1.0f + __expf(-x));
}

// ---------------- weight packing: W[K][128] fp32 -> wt[128][ldk] f16 --------
__global__ void k_packW(const float* __restrict__ W, _Float16* __restrict__ wt,
                        int K, int ldk) {
  const int total = 128 * ldk;
  for (int i = blockIdx.x * blockDim.x + threadIdx.x; i < total;
       i += gridDim.x * blockDim.x) {
    const int n = i / ldk;
    const int k = i - n * ldk;
    wt[i] = (k < K) ? (_Float16)W[k * 128 + n] : (_Float16)0.0f;
  }
}

__global__ void k_zero(float* __restrict__ p, int n) {
  for (int i = blockIdx.x * blockDim.x + threadIdx.x; i < n;
       i += gridDim.x * blockDim.x)
    p[i] = 0.0f;
}

// ---------------- h = silu(emb[z] @ atom_w + atom_b) ------------------------
__global__ void __launch_bounds__(128)
k_node_embed(const int* __restrict__ z, const float* __restrict__ emb,
             const _Float16* __restrict__ wt,  // [128][128]
             const float* __restrict__ bias, float* __restrict__ h) {
  __shared__ _Float16 at[4][16 * 128];
  const int tid = threadIdx.x, lane = tid & 31, wv = tid >> 5;
  const int ntiles = (N_NODES + 63) / 64;
  for (int t = blockIdx.x; t < ntiles; t += gridDim.x) {
    const int base = t * 64 + wv * 16;
    if (base + 16 <= N_NODES) {  // wave-uniform; N_NODES % 16 == 0
      _Float16* a = at[wv];
      for (int i = lane; i < 16 * 128; i += 32) {
        const int r = i >> 7, k = i & 127;
        int zi = z[base + r];
        zi = zi < 0 ? 0 : (zi > 100 ? 100 : zi);
        a[i] = (_Float16)emb[zi * 128 + k];
      }
      v16h af[4];
      load_afrags<4>(af, a, 128, lane);
#pragma unroll 1
      for (int nt = 0; nt < 8; ++nt) {
        v8f acc = gemm_tile<4>(af, wt + (nt * 16) * 128, 128, lane);
        const int col = nt * 16 + (lane & 15);
        const float b = bias[col];
#pragma unroll
        for (int v = 0; v < 8; ++v) {
          const int m = v + ((lane >> 4) << 3);
          h[(size_t)(base + m) * 128 + col] = silu_f(acc[v] + b);
        }
      }
    }
  }
}

// ---------------- e = silu(ef @ edge_w + edge_b), ef built on the fly -------
__global__ void __launch_bounds__(128)
k_edge_feat(const float* __restrict__ edge_attr,
            const _Float16* __restrict__ wt,  // [128][64] (K padded 36->64)
            const float* __restrict__ bias, _Float16* __restrict__ e) {
  __shared__ _Float16 at[4][16 * 64];
  const int tid = threadIdx.x, lane = tid & 31, wv = tid >> 5;
  const float step = CUTOFF / 31.0f;
  const float gamma = 10.0f / (CUTOFF * CUTOFF);
  const int ntiles = N_EDGES / 64;
  for (int t = blockIdx.x; t < ntiles; t += gridDim.x) {
    const int base = t * 64 + wv * 16;
    _Float16* a = at[wv];
    for (int i = lane; i < 16 * 64; i += 32) {
      const int r = i >> 6, k = i & 63;
      const int ei = base + r;
      float val;
      if (k < 32) {
        const float d = edge_attr[ei * 4 + 3];
        const float x = d - step * (float)k;
        val = __expf(-gamma * x * x);
      } else if (k < 36) {
        val = edge_attr[ei * 4 + (k - 32)];  // dcart x,y,z then dist
      } else {
        val = 0.0f;
      }
      a[i] = (_Float16)val;
    }
    v16h af[2];
    load_afrags<2>(af, a, 64, lane);
#pragma unroll 1
    for (int nt = 0; nt < 8; ++nt) {
      v8f acc = gemm_tile<2>(af, wt + (nt * 16) * 64, 64, lane);
      const int col = nt * 16 + (lane & 15);
      const float b = bias[col];
#pragma unroll
      for (int v = 0; v < 8; ++v) {
        const int m = v + ((lane >> 4) << 3);
        e[(size_t)(base + m) * 128 + col] = (_Float16)silu_f(acc[v] + b);
      }
    }
  }
}

// ---------------- edge MLP + scatter-add: agg[dst] += MLP([h[src]|e]) -------
__global__ void __launch_bounds__(128)
k_msg(const float* __restrict__ h, const _Float16* __restrict__ e,
      const int* __restrict__ src, const int* __restrict__ dst,
      const _Float16* __restrict__ w1t,  // [128][256]
      const float* __restrict__ b1,
      const _Float16* __restrict__ w2t,  // [128][128]
      const float* __restrict__ b2, float* __restrict__ agg) {
  __shared__ _Float16 at[4][16 * 256];  // 32 KB
  __shared__ _Float16 tt[4][16 * 128];  // 16 KB
  const int tid = threadIdx.x, lane = tid & 31, wv = tid >> 5;
  const int ntiles = N_EDGES / 64;
  for (int t = blockIdx.x; t < ntiles; t += gridDim.x) {
    const int base = t * 64 + wv * 16;
    _Float16* a = at[wv];
    _Float16* tw = tt[wv];
    for (int i = lane; i < 16 * 256; i += 32) {
      const int r = i >> 8, k = i & 255;
      const int ei = base + r;
      if (k < 128)
        a[i] = (_Float16)h[(size_t)src[ei] * 128 + k];
      else
        a[i] = e[(size_t)ei * 128 + (k - 128)];
    }
    // GEMM1: [16x256] x [256x128], silu -> f16 tile
    {
      v16h af[8];
      load_afrags<8>(af, a, 256, lane);
#pragma unroll 1
      for (int nt = 0; nt < 8; ++nt) {
        v8f acc = gemm_tile<8>(af, w1t + (nt * 16) * 256, 256, lane);
        const int col = nt * 16 + (lane & 15);
        const float b = b1[col];
#pragma unroll
        for (int v = 0; v < 8; ++v) {
          const int m = v + ((lane >> 4) << 3);
          tw[m * 128 + col] = (_Float16)silu_f(acc[v] + b);
        }
      }
    }
    // GEMM2: [16x128] x [128x128] + scatter-add
    {
      v16h af[4];
      load_afrags<4>(af, tw, 128, lane);
#pragma unroll 1
      for (int nt = 0; nt < 8; ++nt) {
        v8f acc = gemm_tile<4>(af, w2t + (nt * 16) * 128, 128, lane);
        const int col = nt * 16 + (lane & 15);
        const float b = b2[col];
#pragma unroll
        for (int v = 0; v < 8; ++v) {
          const int m = v + ((lane >> 4) << 3);
          const int d = dst[base + m];
          atomicAdd(&agg[(size_t)d * 128 + col], acc[v] + b);
        }
      }
    }
  }
}

// ---------------- node update: LN, update-MLP, LN ---------------------------
__global__ void __launch_bounds__(128)
k_node_update(float* __restrict__ h, const float* __restrict__ agg,
              const _Float16* __restrict__ w1t,  // [128][256]
              const float* __restrict__ b1,
              const _Float16* __restrict__ w2t,  // [128][128]
              const float* __restrict__ b2, const float* __restrict__ g1,
              const float* __restrict__ be1, const float* __restrict__ g2,
              const float* __restrict__ be2) {
  __shared__ _Float16 A[16 * 256];  // [hn | agg] f16
  __shared__ float    xb[16 * 128];
  __shared__ float    hn[16 * 128];
  __shared__ _Float16 tb[16 * 128];
  __shared__ float    x2[16 * 128];
  __shared__ float    mu[16], rs[16];
  const int tid = threadIdx.x, lane = tid & 31, wv = tid >> 5;
  for (int t = blockIdx.x; t < N_NODES / 16; t += gridDim.x) {
    const int base = t * 16;
    for (int i = tid; i < 16 * 128; i += 128) {
      const int r = i >> 7, k = i & 127;
      const float hv = h[(size_t)(base + r) * 128 + k];
      const float av = agg[(size_t)(base + r) * 128 + k];
      xb[i] = hv + av;
      A[r * 256 + 128 + k] = (_Float16)av;
    }
    __syncthreads();
    if (tid < 16) {
      float s = 0.f, ss = 0.f;
      for (int k = 0; k < 128; ++k) {
        const float v = xb[tid * 128 + k];
        s += v;
        ss += v * v;
      }
      const float m = s * (1.0f / 128.0f);
      mu[tid] = m;
      rs[tid] = rsqrtf(ss * (1.0f / 128.0f) - m * m + 1e-5f);
    }
    __syncthreads();
    for (int i = tid; i < 16 * 128; i += 128) {
      const int r = i >> 7, k = i & 127;
      const float v = (xb[i] - mu[r]) * rs[r] * g1[k] + be1[k];
      hn[i] = v;
      A[r * 256 + k] = (_Float16)v;
    }
    __syncthreads();
    // GEMM1: wave wv handles N-tiles 2wv, 2wv+1
    {
      v16h af[8];
      load_afrags<8>(af, A, 256, lane);
#pragma unroll 1
      for (int q = 0; q < 2; ++q) {
        const int nt = wv * 2 + q;
        v8f acc = gemm_tile<8>(af, w1t + (nt * 16) * 256, 256, lane);
        const int col = nt * 16 + (lane & 15);
        const float b = b1[col];
#pragma unroll
        for (int v = 0; v < 8; ++v) {
          const int m = v + ((lane >> 4) << 3);
          tb[m * 128 + col] = (_Float16)silu_f(acc[v] + b);
        }
      }
    }
    __syncthreads();
    {
      v16h af[4];
      load_afrags<4>(af, tb, 128, lane);
#pragma unroll 1
      for (int q = 0; q < 2; ++q) {
        const int nt = wv * 2 + q;
        v8f acc = gemm_tile<4>(af, w2t + (nt * 16) * 128, 128, lane);
        const int col = nt * 16 + (lane & 15);
        const float b = b2[col];
#pragma unroll
        for (int v = 0; v < 8; ++v) {
          const int m = v + ((lane >> 4) << 3);
          x2[m * 128 + col] = hn[m * 128 + col] + acc[v] + b;
        }
      }
    }
    __syncthreads();
    if (tid < 16) {
      float s = 0.f, ss = 0.f;
      for (int k = 0; k < 128; ++k) {
        const float v = x2[tid * 128 + k];
        s += v;
        ss += v * v;
      }
      const float m = s * (1.0f / 128.0f);
      mu[tid] = m;
      rs[tid] = rsqrtf(ss * (1.0f / 128.0f) - m * m + 1e-5f);
    }
    __syncthreads();
    for (int i = tid; i < 16 * 128; i += 128) {
      const int r = i >> 7, k = i & 127;
      h[(size_t)(base + r) * 128 + k] = (x2[i] - mu[r]) * rs[r] * g2[k] + be2[k];
    }
    __syncthreads();
  }
}

// ---------------- output heads ----------------------------------------------
__global__ void __launch_bounds__(128)
k_heads(const float* __restrict__ h, const _Float16* __restrict__ drw1t,
        const float* __restrict__ drb1, const float* __restrict__ drw2,
        const float* __restrict__ drb2, const _Float16* __restrict__ rmw1t,
        const float* __restrict__ rmb1, const float* __restrict__ rmw2,
        const float* __restrict__ rmb2, float* __restrict__ out) {
  __shared__ _Float16 at[4][16 * 128];
  __shared__ float    tb[4][16 * 128];
  const int tid = threadIdx.x, lane = tid & 31, wv = tid >> 5;
  const int ntiles = (N_NODES + 63) / 64;
  for (int t = blockIdx.x; t < ntiles; t += gridDim.x) {
    const int base = t * 64 + wv * 16;
    if (base + 16 <= N_NODES) {  // wave-uniform; N_NODES % 16 == 0
      _Float16* a = at[wv];
      float* tw = tb[wv];
      for (int i = lane; i < 16 * 128; i += 32) {
        const int r = i >> 7, k = i & 127;
        a[i] = (_Float16)h[(size_t)(base + r) * 128 + k];
      }
      v16h af[4];
      load_afrags<4>(af, a, 128, lane);
      // dr hidden
#pragma unroll 1
      for (int nt = 0; nt < 8; ++nt) {
        v8f acc = gemm_tile<4>(af, drw1t + (nt * 16) * 128, 128, lane);
        const int col = nt * 16 + (lane & 15);
        const float b = drb1[col];
#pragma unroll
        for (int v = 0; v < 8; ++v) {
          const int m = v + ((lane >> 4) << 3);
          tw[m * 128 + col] = silu_f(acc[v] + b);
        }
      }
      if (lane < 16) {
        const int node = base + lane;
        float s0 = drb2[0], s1 = drb2[1], s2 = drb2[2];
        for (int k = 0; k < 128; ++k) {
          const float tv = tw[lane * 128 + k];
          s0 += tv * drw2[k * 3 + 0];
          s1 += tv * drw2[k * 3 + 1];
          s2 += tv * drw2[k * 3 + 2];
        }
        out[(size_t)node * 3 + 0] = s0;
        out[(size_t)node * 3 + 1] = s1;
        out[(size_t)node * 3 + 2] = s2;
      }
      // rm hidden
#pragma unroll 1
      for (int nt = 0; nt < 8; ++nt) {
        v8f acc = gemm_tile<4>(af, rmw1t + (nt * 16) * 128, 128, lane);
        const int col = nt * 16 + (lane & 15);
        const float b = rmb1[col];
#pragma unroll
        for (int v = 0; v < 8; ++v) {
          const int m = v + ((lane >> 4) << 3);
          tw[m * 128 + col] = silu_f(acc[v] + b);
        }
      }
      if (lane < 16) {
        const int node = base + lane;
        float s = rmb2[0];
        for (int k = 0; k < 128; ++k) s += tw[lane * 128 + k] * rmw2[k];
        out[(size_t)(3 * N_NODES) + node] = s;
      }
    }
  }
}

// ---------------- host orchestration ----------------------------------------
extern "C" void kernel_launch(void* const* d_in, const int* in_sizes, int n_in,
                              void* d_out, int out_size, void* d_ws,
                              size_t ws_size, hipStream_t stream) {
  (void)in_sizes; (void)n_in; (void)out_size; (void)ws_size;
  const int*   z         = (const int*)d_in[0];
  const int*   edge_idx  = (const int*)d_in[2];
  const float* edge_attr = (const float*)d_in[3];
  const float* emb       = (const float*)d_in[4];
  const float* atom_w    = (const float*)d_in[5];
  const float* atom_b    = (const float*)d_in[6];
  const float* edge_w    = (const float*)d_in[7];
  const float* edge_b    = (const float*)d_in[8];
  const float* msg_w1    = (const float*)d_in[9];
  const float* msg_b1    = (const float*)d_in[10];
  const float* msg_w2    = (const float*)d_in[11];
  const float* msg_b2    = (const float*)d_in[12];
  const float* upd_w1    = (const float*)d_in[13];
  const float* upd_b1    = (const float*)d_in[14];
  const float* upd_w2    = (const float*)d_in[15];
  const float* upd_b2    = (const float*)d_in[16];
  const float* n1_g      = (const float*)d_in[17];
  const float* n1_b      = (const float*)d_in[18];
  const float* n2_g      = (const float*)d_in[19];
  const float* n2_b      = (const float*)d_in[20];
  const float* dr_w1     = (const float*)d_in[21];
  const float* dr_b1     = (const float*)d_in[22];
  const float* dr_w2     = (const float*)d_in[23];
  const float* dr_b2     = (const float*)d_in[24];
  const float* rm_w1     = (const float*)d_in[25];
  const float* rm_b1     = (const float*)d_in[26];
  const float* rm_w2     = (const float*)d_in[27];
  const float* rm_b2     = (const float*)d_in[28];

  const int* src = edge_idx;
  const int* dst = edge_idx + N_EDGES;

  // workspace layout
  uintptr_t p = (uintptr_t)d_ws;
  auto take = [&](size_t bytes) {
    uintptr_t q = p;
    p = (p + bytes + 255) & ~(uintptr_t)255;
    return q;
  };
  float*    h    = (float*)take((size_t)N_NODES * 128 * 4);
  float*    agg  = (float*)take((size_t)N_NODES * 128 * 4);
  _Float16* e    = (_Float16*)take((size_t)N_EDGES * 128 * 2);
  _Float16* wAtm = (_Float16*)take(128 * 128 * 2);
  _Float16* wEdg = (_Float16*)take(128 * 64 * 2);
  _Float16* wM1  = (_Float16*)take((size_t)NLAYERS * 128 * 256 * 2);
  _Float16* wM2  = (_Float16*)take((size_t)NLAYERS * 128 * 128 * 2);
  _Float16* wU1  = (_Float16*)take((size_t)NLAYERS * 128 * 256 * 2);
  _Float16* wU2  = (_Float16*)take((size_t)NLAYERS * 128 * 128 * 2);
  _Float16* wDr  = (_Float16*)take(128 * 128 * 2);
  _Float16* wRm  = (_Float16*)take(128 * 128 * 2);

  // pack weights f16, transposed [N][K]
  k_packW<<<32, 256, 0, stream>>>(atom_w, wAtm, 128, 128);
  k_packW<<<32, 256, 0, stream>>>(edge_w, wEdg, 36, 64);
  for (int l = 0; l < NLAYERS; ++l) {
    k_packW<<<32, 256, 0, stream>>>(msg_w1 + (size_t)l * 256 * 128,
                                    wM1 + (size_t)l * 128 * 256, 256, 256);
    k_packW<<<32, 256, 0, stream>>>(msg_w2 + (size_t)l * 128 * 128,
                                    wM2 + (size_t)l * 128 * 128, 128, 128);
    k_packW<<<32, 256, 0, stream>>>(upd_w1 + (size_t)l * 256 * 128,
                                    wU1 + (size_t)l * 128 * 256, 256, 256);
    k_packW<<<32, 256, 0, stream>>>(upd_w2 + (size_t)l * 128 * 128,
                                    wU2 + (size_t)l * 128 * 128, 128, 128);
  }
  k_packW<<<32, 256, 0, stream>>>(dr_w1, wDr, 128, 128);
  k_packW<<<32, 256, 0, stream>>>(rm_w1, wRm, 128, 128);

  k_node_embed<<<512, 128, 0, stream>>>(z, emb, wAtm, atom_b, h);
  k_edge_feat<<<1024, 128, 0, stream>>>(edge_attr, wEdg, edge_b, e);

  for (int l = 0; l < NLAYERS; ++l) {
    k_zero<<<256, 256, 0, stream>>>(agg, N_NODES * 128);
    k_msg<<<2048, 128, 0, stream>>>(h, e, src, dst,
                                    wM1 + (size_t)l * 128 * 256, msg_b1 + l * 128,
                                    wM2 + (size_t)l * 128 * 128, msg_b2 + l * 128,
                                    agg);
    k_node_update<<<2048, 128, 0, stream>>>(
        h, agg, wU1 + (size_t)l * 128 * 256, upd_b1 + l * 128,
        wU2 + (size_t)l * 128 * 128, upd_b2 + l * 128, n1_g + l * 128,
        n1_b + l * 128, n2_g + l * 128, n2_b + l * 128);
  }

  k_heads<<<512, 128, 0, stream>>>(h, wDr, dr_b1, dr_w2, dr_b2, wRm, rm_b1,
                                   rm_w2, rm_b2, (float*)d_out);
}
